// Net_32229434589761
// MI455X (gfx1250) — compile-verified, matrix operand order checked by reading, not run on hardware
//
#include <hip/hip_runtime.h>

// ---------------- CDNA5 WMMA types ----------------
typedef _Float16 v16h __attribute__((ext_vector_type(16)));
typedef float    v8f  __attribute__((ext_vector_type(8)));

#define N_NODES 50000
#define N_EDGES 400000
#define CDIV(a, b) (((a) + (b) - 1) / (b))

#define KC 320        // K-chunk staged in LDS (multiple of 32)
#define BS_STRIDE 328 // halves; 656B row stride => 2-way-minimal ds_load_b128 banking

// Load one 16-lane-pattern fragment (16 f16 = 32B) as two b128 loads.
// ISA 7.12.2 16-bit A/B layout: lane<16 holds K{+0..7, +16..23} of its row,
// lane>=16 holds K{+8..15, +24..31}. Caller bakes the +half*8 into p.
// Works for both global and LDS pointers (addrspace inferred -> b128 / ds b128).
__device__ __forceinline__ v16h load_frag(const _Float16* p) {
    union { uint4 u[2]; v16h v; } f;
    f.u[0] = *(const uint4*)(p);
    f.u[1] = *(const uint4*)(p + 16);
    return f.v;
}

// ---------------- GEMM: C[M,N] = A[M,K] * Bt[N,K]^T  (f16 in, f32 out) ------
// Workgroup = 8 waves => 128(M) x 64(N) tile. B strip staged in LDS in KC
// chunks (8x reuse across waves); each wave owns 16 rows, 4 accumulators.
__global__ __launch_bounds__(256) void k_gemm_f16(
    const _Float16* __restrict__ A, const _Float16* __restrict__ Bt,
    float* __restrict__ C, int Mtiles, int NT64, int Kp,
    int lda, int ldb, int ldc) {
    __shared__ _Float16 Bs[64 * BS_STRIDE];  // 41.98 KB

    int blocksM = (Mtiles + 7) >> 3;
    int bm = blockIdx.x % blocksM;
    int nt = blockIdx.x / blocksM;
    int wid = threadIdx.x >> 5;
    int mt = bm * 8 + wid;                 // this wave's 16-row M tile
    int lane = threadIdx.x & 31;
    int hf = lane >> 4;                    // which K-half pattern this lane owns
    int r  = lane & 15;
    bool live = (mt < Mtiles);             // wave-uniform (EXEC all-ones inside)

    const _Float16* arow = A + (size_t)(mt * 16 + r) * lda + hf * 8;
    const _Float16* bsrc = Bt + (size_t)(nt * 64) * ldb;
    const _Float16* bl = &Bs[r * BS_STRIDE + hf * 8];

    v8f c0 = {}, c1 = {}, c2 = {}, c3 = {};
    for (int kc0 = 0; kc0 < Kp; kc0 += KC) {
        int Kc = Kp - kc0; if (Kc > KC) Kc = KC;   // 320 or 288, mult of 32
        __syncthreads();                            // Bs safe to overwrite
        // cooperative global -> LDS stage of the 64 x Kc B strip
        int kv = Kc >> 3;                           // 16B vectors per row
        for (int idx = threadIdx.x; idx < 64 * kv; idx += 256) {
            int n  = idx / kv;
            int ko = (idx - n * kv) << 3;
            *(uint4*)(&Bs[n * BS_STRIDE + ko]) =
                *(const uint4*)(bsrc + (size_t)n * ldb + kc0 + ko);
        }
        __syncthreads();
        if (live) {
            int kst = Kc >> 5;
            for (int ks = 0; ks < kst; ++ks) {
                int kg = kc0 + ks * 32;             // global K offset (A)
                int kl = ks * 32;                   // LDS K offset (B)
                __builtin_prefetch(arow + kg + 64, 0, 0);  // global_prefetch_b8
                v16h a  = load_frag(arow + kg);
                v16h b0 = load_frag(bl + kl);
                v16h b1 = load_frag(bl + 16 * BS_STRIDE + kl);
                v16h b2 = load_frag(bl + 32 * BS_STRIDE + kl);
                v16h b3 = load_frag(bl + 48 * BS_STRIDE + kl);
                c0 = __builtin_amdgcn_wmma_f32_16x16x32_f16(false, a, false, b0, (short)0, c0, false, false);
                c1 = __builtin_amdgcn_wmma_f32_16x16x32_f16(false, a, false, b1, (short)0, c1, false, false);
                c2 = __builtin_amdgcn_wmma_f32_16x16x32_f16(false, a, false, b2, (short)0, c2, false, false);
                c3 = __builtin_amdgcn_wmma_f32_16x16x32_f16(false, a, false, b3, (short)0, c3, false, false);
            }
        }
    }
    if (!live) return;
    // C layout: VGPR rr -> row (hf*8 + rr), lane r -> column.
    float* out = C + (size_t)(mt * 16 + hf * 8) * ldc + (size_t)nt * 64 + r;
#pragma unroll
    for (int rr = 0; rr < 8; ++rr) out[(size_t)rr * ldc + 0]  = c0[rr];
#pragma unroll
    for (int rr = 0; rr < 8; ++rr) out[(size_t)rr * ldc + 16] = c1[rr];
#pragma unroll
    for (int rr = 0; rr < 8; ++rr) out[(size_t)rr * ldc + 32] = c2[rr];
#pragma unroll
    for (int rr = 0; rr < 8; ++rr) out[(size_t)rr * ldc + 48] = c3[rr];
}

// ---------------- degree / normalization ----------------
__global__ void k_deg_init(float* deg, int n) {
    int i = blockIdx.x * blockDim.x + threadIdx.x;
    if (i < n) deg[i] = 1.0f;  // self-loop
}
__global__ void k_deg_count(float* deg, const int* __restrict__ dst, int e) {
    int i = blockIdx.x * blockDim.x + threadIdx.x;
    if (i < e) atomicAdd(&deg[dst[i]], 1.0f);
}
__global__ void k_dinv(float* deg, int n) {
    int i = blockIdx.x * blockDim.x + threadIdx.x;
    if (i < n) deg[i] = rsqrtf(deg[i]);  // deg >= 1 always
}

// ---------------- f16 packing (with zero padding) ----------------
__global__ void k_pack_x(_Float16* __restrict__ xb, const float* __restrict__ x,
                         int n, int K, int Kp) {
    long idx = (long)blockIdx.x * blockDim.x + threadIdx.x;
    if (idx >= (long)n * Kp) return;
    int  k = (int)(idx % Kp);
    long i = idx / Kp;
    xb[idx] = (k < K) ? (_Float16)x[i * K + k] : (_Float16)0.0f;
}
// Wt[n][k] = W[k][n], padded with zeros to [Np, Kp]
__global__ void k_pack_wt(_Float16* __restrict__ Wt, const float* __restrict__ W,
                          int K, int Nw, int Kp, int Np) {
    long idx = (long)blockIdx.x * blockDim.x + threadIdx.x;
    if (idx >= (long)Np * Kp) return;
    int  k  = (int)(idx % Kp);
    int  nn = (int)(idx / Kp);
    float v = (k < K && nn < Nw) ? W[(size_t)k * Nw + nn] : 0.0f;
    Wt[idx] = (_Float16)v;
}

// ---------------- GCN aggregation ----------------
// Self-loop term written directly (no atomics): out[i] = dinv[i]^2 * h[i]
__global__ void k_selfloop(float* __restrict__ out, const float* __restrict__ h,
                           const float* __restrict__ dinv, int n, int F,
                           int ldh, int ldo) {
    long idx = (long)blockIdx.x * blockDim.x + threadIdx.x;
    if (idx >= (long)n * F) return;
    int  f = (int)(idx % F);
    long i = idx / F;
    float di = dinv[i];
    out[i * (long)ldo + f] = di * di * h[i * (long)ldh + f];
}
// One block per edge; threads stride over features.
__global__ void k_scatter(float* __restrict__ out, const float* __restrict__ h,
                          const float* __restrict__ dinv,
                          const int* __restrict__ src, const int* __restrict__ dst,
                          int F, int ldh, int ldo) {
    int e = blockIdx.x;
    int s = src[e], d = dst[e];
    float nrm = dinv[s] * dinv[d];
    const float* hs = h + (size_t)s * ldh;
    float* od = out + (size_t)d * ldo;
    for (int f = threadIdx.x; f < F; f += blockDim.x)
        atomicAdd(&od[f], nrm * hs[f]);
}

// bias + relu + f16 pack (pads F -> Fp with zeros)
__global__ void k_relu_f16(_Float16* __restrict__ hb, const float* __restrict__ ha,
                           const float* __restrict__ b, int n, int F, int Fp) {
    long idx = (long)blockIdx.x * blockDim.x + threadIdx.x;
    if (idx >= (long)n * Fp) return;
    int  f = (int)(idx % Fp);
    long i = idx / Fp;
    float v = (f < F) ? fmaxf(ha[i * (long)F + f] + b[f], 0.0f) : 0.0f;
    hb[idx] = (_Float16)v;
}

__global__ void k_bias(float* __restrict__ za, const float* __restrict__ b, int n, int F) {
    long idx = (long)blockIdx.x * blockDim.x + threadIdx.x;
    if (idx >= (long)n * F) return;
    za[idx] += b[(int)(idx % F)];
}

// decode: logits[e] = dot(z[src], z[dst]) over 64 feats (rows 256B-aligned)
__global__ void k_decode(float* __restrict__ logits, const float* __restrict__ z,
                         const int* __restrict__ src, const int* __restrict__ dst, int e) {
    int i = blockIdx.x * blockDim.x + threadIdx.x;
    if (i >= e) return;
    const float4* a4 = (const float4*)(z + (size_t)src[i] * 64);
    const float4* b4 = (const float4*)(z + (size_t)dst[i] * 64);
    float acc = 0.0f;
#pragma unroll
    for (int j = 0; j < 16; ++j) {
        float4 a = a4[j], b = b4[j];
        acc += a.x * b.x + a.y * b.y + a.z * b.z + a.w * b.w;
    }
    logits[i] = acc;
}

// ---------------- host orchestration ----------------
extern "C" void kernel_launch(void* const* d_in, const int* in_sizes, int n_in,
                              void* d_out, int out_size, void* d_ws, size_t ws_size,
                              hipStream_t stream) {
    const float* x  = (const float*)d_in[0];
    const int*   ei = (const int*)d_in[1];
    const float* W1 = (const float*)d_in[2];
    const float* b1 = (const float*)d_in[3];
    const float* W2 = (const float*)d_in[4];
    const float* b2 = (const float*)d_in[5];
    float* logits = (float*)d_out;
    (void)in_sizes; (void)n_in; (void)out_size; (void)ws_size;

    const int N = N_NODES, E = N_EDGES;
    const int K1 = 600, F1 = 628, F2 = 64;
    const int K1p = 608, F1p = 640;          // padded to multiples of 32 (K) / 64 (N)
    const int* src = ei;
    const int* dst = ei + E;

    // ---- workspace layout (lifetime-aliased) ----
    char* ws = (char*)d_ws;
    size_t off = 0;
    auto alloc = [&](size_t bytes) -> char* {
        char* p = ws + off;
        off = (off + bytes + 255) & ~(size_t)255;
        return p;
    };
    float* dinv = (float*)alloc((size_t)N * 4);
    char* slotA = alloc((size_t)N * F1p * 2);         // 64 MB: xb, later h1b
    _Float16* xb   = (_Float16*)slotA;                // [N, 608]
    _Float16* h1b  = (_Float16*)slotA;                // [N, 640] (after xb dead)
    _Float16* W1bt = (_Float16*)alloc((size_t)F1p * K1p * 2);  // [640, 608]
    _Float16* W2bt = (_Float16*)alloc((size_t)F2 * F1p * 2);   // [64, 640]
    char* slotB = alloc((size_t)N * F1p * 4);         // 128 MB: h1, later z+za
    float* h1 = (float*)slotB;                        // [N, 640]
    float* z  = (float*)slotB;                        // [N, 64] (after h1 dead)
    float* za = (float*)(slotB + (size_t)N * F2 * 4); // [N, 64]
    float* h1a = (float*)alloc((size_t)N * F1 * 4);   // [N, 628]

    // ---- normalization: deg -> 1/sqrt(deg) ----
    k_deg_init<<<CDIV(N, 256), 256, 0, stream>>>(dinv, N);
    k_deg_count<<<CDIV(E, 256), 256, 0, stream>>>(dinv, dst, E);
    k_dinv<<<CDIV(N, 256), 256, 0, stream>>>(dinv, N);

    // ---- pack operands to f16 ----
    k_pack_x<<<CDIV((long)N * K1p, 256), 256, 0, stream>>>(xb, x, N, K1, K1p);
    k_pack_wt<<<CDIV((long)F1p * K1p, 256), 256, 0, stream>>>(W1bt, W1, K1, F1, K1p, F1p);
    k_pack_wt<<<CDIV((long)F2 * F1p, 256), 256, 0, stream>>>(W2bt, W2, F1, F2, F1p, F2);

    // ---- layer 1: h1 = x @ W1 (WMMA), aggregate, +b1, relu, repack f16 ----
    {
        int mt = N / 16, nt = F1p / 64;               // 3125, 10
        int blocksM = CDIV(mt, 8);                    // 391
        k_gemm_f16<<<blocksM * nt, 256, 0, stream>>>(
            xb, W1bt, h1, mt, nt, K1p, K1p, K1p, F1p);
    }
    k_selfloop<<<CDIV((long)N * F1, 256), 256, 0, stream>>>(h1a, h1, dinv, N, F1, F1p, F1);
    k_scatter<<<E, 256, 0, stream>>>(h1a, h1, dinv, src, dst, F1, F1p, F1);
    k_relu_f16<<<CDIV((long)N * F1p, 256), 256, 0, stream>>>(h1b, h1a, b1, N, F1, F1p);

    // ---- layer 2: z = h @ W2 (WMMA), aggregate, +b2 ----
    {
        int mt = N / 16, nt = 1;                      // 3125, 1
        int blocksM = CDIV(mt, 8);                    // 391
        k_gemm_f16<<<blocksM * nt, 256, 0, stream>>>(
            h1b, W2bt, z, mt, nt, F1p, F1p, F1p, F2);
    }
    k_selfloop<<<CDIV((long)N * F2, 256), 256, 0, stream>>>(za, z, dinv, N, F2, F2, F2);
    k_scatter<<<E, 64, 0, stream>>>(za, z, dinv, src, dst, F2, F2, F2);
    k_bias<<<CDIV((long)N * F2, 256), 256, 0, stream>>>(za, b2, N, F2);

    // ---- decode ----
    k_decode<<<CDIV(E, 256), 256, 0, stream>>>(logits, za, src, dst, E);
}
